// WMSA_6459630813193
// MI455X (gfx1250) — matrix operand order repaired and checked by reading, not compiled, not run on hardware
//
#include <hip/hip_runtime.h>
#include <hip/hip_bf16.h>

// ---------------------------------------------------------------------------
// Fused Swin W-MSA for gfx1250 (MI455X).
// One workgroup (8 wave32) per window; qkv / scores / probs / attn-out all
// live in the 320KB WGP LDS; every matmul is v_wmma_f32_16x16x32_f16.
// ---------------------------------------------------------------------------

typedef __attribute__((ext_vector_type(8)))  _Float16 h8;
typedef __attribute__((ext_vector_type(16))) _Float16 v16h;
typedef __attribute__((ext_vector_type(8)))  float    v8f;

#define NWIN   8192
#define NTOK   64
#define CDIM   180
#define NHEAD  6
#define HD     30
#define HDP    32          // head dim padded to 32
#define KP     192         // C padded to 192 (6 K-chunks of 32)
#define NQKV   576         // 3 * NHEAD * HDP packed output channels
#define NWIMG  1024

// workspace byte offsets (prep-built constants, ~0.4 MB total)
#define OFF_WQKV   0u
#define LEN_WQKV   (NQKV * KP * 2)                 // f16 [576][192], B-layout
#define OFF_WPROJ  (OFF_WQKV + LEN_WQKV)
#define LEN_WPROJ  (KP * KP * 2)                   // f16 [192][192], B-layout
#define OFF_BQKV   (OFF_WPROJ + LEN_WPROJ)
#define LEN_BQKV   (NQKV * 4)                      // f32 [576]
#define OFF_BIAS   (OFF_BQKV + LEN_BQKV)
#define LEN_BIAS   (NHEAD * 64 * 64 * 4)           // f32 [6][64][64]

// LDS layout (bytes)
#define SM_X     0            // f16 [64][192]  x, later attn-out (reused)
#define SM_QKV   24576        // per head 12288B: q[64][32], k[64][32], vT[32][64]
#define SM_S     98304        // f32 [6][64][64] scores
#define SM_P     196608       // f16 [6][64][64] probs
#define SM_TOTAL 245760

static __device__ __forceinline__ v16h cat16(h8 a, h8 b) {
  return __builtin_shufflevector(a, b, 0,1,2,3,4,5,6,7,8,9,10,11,12,13,14,15);
}
static __device__ __forceinline__ v8f wmma_f16(v16h a, v16h b, v8f c) {
  // D = A(16x32 f16) * B(32x16 f16) + C(16x16 f32)
  return __builtin_amdgcn_wmma_f32_16x16x32_f16(false, a, false, b,
                                                (short)0, c, false, false);
}
// A-fragment: row-major [rows][strideH] f16; lane row = l%16,
// K runs [8*(l/16), +8) and [16+8*(l/16), +8) within chunk kc.
static __device__ __forceinline__ v16h loadA(const _Float16* base, int row,
                                             int strideH, int kc, int ln16, int h2) {
  const _Float16* p = base + (size_t)row * strideH + kc * 32 + 8 * h2;
  return cat16(*(const h8*)p, *(const h8*)(p + 16));
}
// B-fragment: source stored as [n][K] row-major f16; lane col = l%16,
// K runs [8*(l/16), +8) and [16+8*(l/16), +8) within chunk kc.
static __device__ __forceinline__ v16h loadB(const _Float16* base, int nrow,
                                             int strideH, int kc, int h2) {
  const _Float16* p = base + (size_t)nrow * strideH + kc * 32 + 8 * h2;
  return cat16(*(const h8*)p, *(const h8*)(p + 16));
}

// ---------------------------------------------------------------------------
// Prep: padded f16 weights (scale folded into Wq/bq), padded proj weight with
// head-padding K remap, relative-position bias table [H][64][64].
// ---------------------------------------------------------------------------
__global__ void wmsa_prep(const float* __restrict__ qkv_w,
                          const float* __restrict__ qkv_b,
                          const float* __restrict__ proj_w,
                          const float* __restrict__ rpb,
                          char* __restrict__ ws) {
  const float scale = 0.18257418583505536f;   // 30^-0.5
  int idx = blockIdx.x * blockDim.x + threadIdx.x;

  _Float16* wqkv  = (_Float16*)(ws + OFF_WQKV);
  _Float16* wproj = (_Float16*)(ws + OFF_WPROJ);
  float*    bqkv  = (float*)   (ws + OFF_BQKV);
  float*    bias  = (float*)   (ws + OFF_BIAS);

  if (idx < NQKV * KP) {                      // Wqkv B-layout [n][k]
    int n = idx / KP, c = idx % KP;
    int qi = n / 192, rem = n % 192, h = rem / HDP, d = rem % HDP;
    float v = 0.f;
    if (d < HD && c < CDIM) {
      v = qkv_w[(size_t)(qi * CDIM + h * HD + d) * CDIM + c];
      if (qi == 0) v *= scale;
    }
    wqkv[idx] = (_Float16)v;
    return;
  }
  idx -= NQKV * KP;
  if (idx < KP * KP) {                        // Wproj B-layout [n][kpad]
    int n = idx / KP, kp = idx % KP, h = kp / HDP, d = kp % HDP;
    float v = (n < CDIM && d < HD) ? proj_w[(size_t)n * CDIM + h * HD + d] : 0.f;
    wproj[idx] = (_Float16)v;
    return;
  }
  idx -= KP * KP;
  if (idx < NQKV) {                           // packed qkv bias
    int qi = idx / 192, rem = idx % 192, h = rem / HDP, d = rem % HDP;
    float v = 0.f;
    if (d < HD) { v = qkv_b[qi * CDIM + h * HD + d]; if (qi == 0) v *= scale; }
    bqkv[idx] = v;
    return;
  }
  idx -= NQKV;
  if (idx < NHEAD * 64 * 64) {                // relative position bias
    int h = idx / 4096, m = (idx >> 6) & 63, n = idx & 63;
    int i0 = m >> 3, j0 = m & 7, i1 = n >> 3, j1 = n & 7;
    int r = (i0 - i1 + 7) * 15 + (j0 - j1 + 7);
    bias[idx] = rpb[r * NHEAD + h];
  }
}

// ---------------------------------------------------------------------------
// Fused per-window kernel: 256 threads = 8 wave32, one window per block.
// ---------------------------------------------------------------------------
__global__ __launch_bounds__(256)
void wmsa_fused(const float* __restrict__ x,
                const float* __restrict__ mask,
                const _Float16* __restrict__ wqkv,
                const float* __restrict__ bqkv,
                const _Float16* __restrict__ wproj,
                const float* __restrict__ projb,
                const float* __restrict__ bias,
                float* __restrict__ out) {
  extern __shared__ __align__(16) char smem[];
  _Float16* sX   = (_Float16*)(smem + SM_X);    // [64][192], later attn-out
  _Float16* sQKV = (_Float16*)(smem + SM_QKV);  // per head: q | k | vT
  float*    sS   = (float*)   (smem + SM_S);    // [6][64][64]
  _Float16* sP   = (_Float16*)(smem + SM_P);    // [6][64][64]

  const int w    = blockIdx.x;
  const int tid  = threadIdx.x;
  const int lane = tid & 31;
  const int wv   = tid >> 5;       // wave 0..7
  const int ln16 = lane & 15;
  const int h2   = lane >> 4;

  // ---- stage x window into LDS as f16, zero-pad c in [180,192) -----------
  {
    const float* xw = x + (size_t)w * NTOK * CDIM;
    for (int i = tid; i < NTOK * KP; i += 256) {
      int m = i / KP, c = i % KP;
      sX[i] = (_Float16)((c < CDIM) ? xw[m * CDIM + c] : 0.f);
    }
  }
  __syncthreads();

  // ---- QKV GEMM: [64,192] x [192,576] -> q/k/vT in LDS --------------------
  for (int nt = wv; nt < NQKV / 16; nt += 8) {           // 36 N-strips
    v16h B[6];
    for (int kc = 0; kc < 6; ++kc)
      B[kc] = loadB(wqkv, nt * 16 + ln16, KP, kc, h2);
    const float bval = bqkv[nt * 16 + ln16];

    const int qi   = nt / 12;                            // 0=q 1=k 2=v
    const int rem  = nt % 12;
    const int hh   = rem >> 1;
    const int d    = (rem & 1) * 16 + ln16;              // head-local dim
    const int isV  = (qi == 2);
    const int qkBase = qi * 2048 + d;                    // + m*32
    const int vBase  = 4096 + d * 64;                    // + m
    _Float16* qh = sQKV + hh * 6144;                     // halves
    for (int mt = 0; mt < 4; ++mt) {
      v8f acc = {};
      for (int kc = 0; kc < 6; ++kc)
        acc = wmma_f16(loadA(sX, mt * 16 + ln16, KP, kc, ln16, h2), B[kc], acc);
      const int m0 = mt * 16 + 8 * h2;
      for (int r = 0; r < 8; ++r) {
        const int m = m0 + r;
        // branchless scatter: q/k row-major [64][32], v transposed [32][64]
        const int addr = isV ? (vBase + m) : (qkBase + m * 32);
        qh[addr] = (_Float16)(acc[r] + bval);
      }
    }
  }
  __syncthreads();

  // ---- scores = q @ k^T : per head [64][64] f32 in LDS --------------------
  for (int s = wv; s < NHEAD * 4; s += 8) {              // 24 column strips
    const int hh = s >> 2, nt = s & 3;
    const _Float16* qh = sQKV + hh * 6144;
    v16h B = loadB(qh + 2048, nt * 16 + ln16, 32, 0, h2);
    float* sh = sS + hh * 4096;
    const int n = nt * 16 + ln16;
    for (int mt = 0; mt < 4; ++mt) {
      v8f acc = {};
      acc = wmma_f16(loadA(qh, mt * 16 + ln16, 32, 0, ln16, h2), B, acc);
      for (int r = 0; r < 8; ++r)
        sh[(mt * 16 + r + 8 * h2) * 64 + n] = acc[r];
    }
  }
  __syncthreads();

  // ---- softmax rows (+ rel-pos bias + window mask), probs -> f16 ----------
  {
    const int mi = w & (NWIMG - 1);
    for (int row = tid; row < NHEAD * 64; row += 256) {
      const int hh = row >> 6, m = row & 63;
      const float* srow = sS + hh * 4096 + m * 64;
      const float* brow = bias + hh * 4096 + m * 64;
      const float* mrow = mask + (size_t)mi * 4096 + m * 64;
      float mx = -3.0e38f;
      for (int n = 0; n < 64; ++n)
        mx = fmaxf(mx, srow[n] + brow[n] + mrow[n]);
      _Float16* prow = sP + hh * 4096 + m * 64;
      float sum = 0.f;
      for (int n = 0; n < 64; ++n) {
        float e = __expf(srow[n] + brow[n] + mrow[n] - mx);
        prow[n] = (_Float16)e;                    // unnormalized (<=1, f16-safe)
        sum += e;
      }
      float inv = 1.f / sum;
      for (int n = 0; n < 64; ++n)
        prow[n] = (_Float16)((float)prow[n] * inv);
    }
  }
  __syncthreads();

  // ---- attn = probs @ v : back into sX (x is dead) ------------------------
  for (int s = wv; s < NHEAD * 2; s += 8) {              // 12 strips
    const int hh = s >> 1, nt = s & 1;
    const _Float16* vt = sQKV + hh * 6144 + 4096;        // vT [32][64]
    v16h B0 = loadB(vt, nt * 16 + ln16, 64, 0, h2);
    v16h B1 = loadB(vt, nt * 16 + ln16, 64, 1, h2);
    const _Float16* ph = sP + hh * 4096;
    const int cc = hh * 32 + nt * 16 + ln16;             // attn column
    for (int mt = 0; mt < 4; ++mt) {
      v8f acc = {};
      acc = wmma_f16(loadA(ph, mt * 16 + ln16, 64, 0, ln16, h2), B0, acc);
      acc = wmma_f16(loadA(ph, mt * 16 + ln16, 64, 1, ln16, h2), B1, acc);
      for (int r = 0; r < 8; ++r)
        sX[(mt * 16 + r + 8 * h2) * KP + cc] = (_Float16)acc[r];
    }
  }
  __syncthreads();

  // ---- proj GEMM: [64,192] x [192,180] + bias -> d_out --------------------
  {
    float* outw = out + (size_t)w * NTOK * CDIM;
    for (int nt = wv; nt < 12; nt += 8) {
      v16h B[6];
      for (int kc = 0; kc < 6; ++kc)
        B[kc] = loadB(wproj, nt * 16 + ln16, KP, kc, h2);
      const int n = nt * 16 + ln16;
      const float bv = (n < CDIM) ? projb[n] : 0.f;
      for (int mt = 0; mt < 4; ++mt) {
        v8f acc = {};
        for (int kc = 0; kc < 6; ++kc)
          acc = wmma_f16(loadA(sX, mt * 16 + ln16, KP, kc, ln16, h2), B[kc], acc);
        if (n < CDIM)
          for (int r = 0; r < 8; ++r)
            outw[(mt * 16 + r + 8 * h2) * CDIM + n] = acc[r] + bv;
      }
    }
  }
}

extern "C" void kernel_launch(void* const* d_in, const int* in_sizes, int n_in,
                              void* d_out, int out_size, void* d_ws, size_t ws_size,
                              hipStream_t stream) {
  (void)in_sizes; (void)n_in; (void)out_size; (void)ws_size;
  const float* x      = (const float*)d_in[0];
  const float* mask   = (const float*)d_in[1];
  const float* qkv_w  = (const float*)d_in[2];
  const float* qkv_b  = (const float*)d_in[3];
  const float* proj_w = (const float*)d_in[4];
  const float* proj_b = (const float*)d_in[5];
  const float* rpb    = (const float*)d_in[6];
  char* ws = (char*)d_ws;

  const int prep_elems = NQKV * KP + KP * KP + NQKV + NHEAD * 64 * 64;
  wmsa_prep<<<(prep_elems + 255) / 256, 256, 0, stream>>>(qkv_w, qkv_b, proj_w,
                                                          rpb, ws);

  wmsa_fused<<<NWIN, 256, SM_TOTAL, stream>>>(
      x, mask,
      (const _Float16*)(ws + OFF_WQKV),
      (const float*)(ws + OFF_BQKV),
      (const _Float16*)(ws + OFF_WPROJ),
      proj_b,
      (const float*)(ws + OFF_BIAS),
      (float*)d_out);
}
